// GraphCritic_62285615727119
// MI455X (gfx1250) — matrix operand order repaired and checked by reading, not compiled, not run on hardware
//
#include <hip/hip_runtime.h>
#include <hip/hip_bf16.h>

// ---------------------------------------------------------------------------
// GraphCritic on MI455X (gfx1250).
// Dominant cost: two adjacency (8192x8192 f32 = 256MB) aggregations.
// HBM floor = 512MB / 23.3 TB/s ~= 22us. Adjacency tiles are converted to
// bf16 in-register and fed to v_wmma_f32_16x16x32_bf16 (f32 accumulate) so
// the matrix engine is never the bottleneck; f32 WMMA (16x16x4) would make
// the 26 GFLOP of aggregation compute-bound at 64 FLOP/byte.
// Aggregation uses a ping-pong LDS pipeline: next chunk's global loads are
// issued into registers, current chunk's ds_load+WMMA hide the latency, then
// cvt+ds_store fills the other buffer -- one barrier per 64-K chunk.
// ---------------------------------------------------------------------------

typedef __attribute__((ext_vector_type(16))) __bf16 v16bf;
typedef __attribute__((ext_vector_type(8)))  float  v8f;

#define GN  8192   // nodes
#define GF  64     // node features
#define GA  16     // action dim
#define GH1 128
#define GH2 64

static __device__ __forceinline__ unsigned pack2_bf16(float a, float b) {
    __bf16 x = (__bf16)a, y = (__bf16)b;
    unsigned short ux = __builtin_bit_cast(unsigned short, x);
    unsigned short uy = __builtin_bit_cast(unsigned short, y);
    return (unsigned)ux | ((unsigned)uy << 16);
}

// ---------------------------------------------------------------------------
// Kernel 1: x1T[h][n] = bf16( relu(nodes @ We1 + be1) ), transposed to K-major
// so the aggregation's WMMA B operand is a contiguous 32B load per lane.
// block = 256 threads = 32 nodes (lanes) x 8 h-groups. grid = 8192/32 = 256.
// ---------------------------------------------------------------------------
__global__ __launch_bounds__(256) void dense1_kernel(
    const float* __restrict__ nodes, const float* __restrict__ We1,
    const float* __restrict__ be1, __bf16* __restrict__ x1T)
{
    __shared__ float sW[GF * GH1];     // 32 KB
    __shared__ float sB[GH1];
    __shared__ float nrow[32][GF + 1]; // padded, conflict-free
    const int tid = threadIdx.x;
    for (int i = tid; i < GF * GH1; i += 256) sW[i] = We1[i];
    for (int i = tid; i < GH1; i += 256) sB[i] = be1[i];
    for (int i = tid; i < 32 * GF; i += 256)
        nrow[i >> 6][i & 63] = nodes[(size_t)(blockIdx.x * 32 + (i >> 6)) * GF + (i & 63)];
    __syncthreads();

    const int lane = tid & 31;
    const int grp  = tid >> 5;                 // 0..7, each covers 16 h
    const int n    = blockIdx.x * 32 + lane;
    #pragma unroll 4
    for (int hh = 0; hh < 16; ++hh) {
        const int h = grp * 16 + hh;
        float s = sB[h];
        #pragma unroll 8
        for (int f = 0; f < GF; ++f) s = fmaf(nrow[lane][f], sW[f * GH1 + h], s);
        s = fmaxf(s, 0.0f);
        x1T[(size_t)h * GN + n] = (__bf16)s;   // coalesced along lanes
    }
}

// ---------------------------------------------------------------------------
// Aggregation: out[N][NC] = adjacency[N][N] @ x   (x given transposed bf16).
// One workgroup = 32 output rows x NC cols; each wave owns two 16x16 C tiles
// (rows lm and lm+16) sharing one B fragment. Adjacency chunks (32 rows x
// 64 K, f32) are staged into ping-pong LDS buffers as packed bf16 pairs in
// natural K order; fragments are read in contiguous K order (k = 16*half + e).
// ---------------------------------------------------------------------------
template <int NC>
__global__ __launch_bounds__((NC / 16) * 32) void agg_kernel(
    const float* __restrict__ adj, const __bf16* __restrict__ xT,
    float* __restrict__ out)
{
    constexpr int NT     = (NC / 16) * 32;   // 256 (NC=128) or 128 (NC=64)
    constexpr int ROWS   = 32;               // M rows per workgroup
    constexpr int TRIPS  = (ROWS * 32) / NT; // staging float2's per thread
    constexpr int RPT    = NT / 32;          // staging rows per trip
    constexpr int NCHUNK = GN / 64;
    // stride 36 uints: 144B rows keep uint4 sub-reads 16B aligned and break
    // systematic LDS bank conflicts.
    __shared__ unsigned A_lds[2][ROWS][36];

    const int tid  = threadIdx.x;
    const int wave = tid >> 5;
    const int lane = tid & 31;
    const int half = lane >> 4;     // 0: lanes 0-15, 1: lanes 16-31
    const int lm   = lane & 15;     // row within A tile / col within B tile
    const int m0   = blockIdx.x * ROWS;
    const int col  = wave * 16 + lm;

    const int sm = tid >> 5;        // staging row
    const int sp = tid & 31;        // staging pair (2 K) index
    const float*  aBase = adj + (size_t)(m0 + sm) * GN + 2 * sp;
    const __bf16* bBase = xT + (size_t)col * GN + 16 * half;

    // prologue: stage chunk 0 into buffer 0 (constant trip count, no masks)
    #pragma unroll
    for (int t = 0; t < TRIPS; ++t) {
        const float2 f = *(const float2*)(aBase + (size_t)(t * RPT) * GN);
        A_lds[0][sm + t * RPT][sp] = pack2_bf16(f.x, f.y);
    }
    __syncthreads();

    v8f acc0 = {}, acc1 = {};
    for (int i = 0; i < NCHUNK; ++i) {
        const int  cur  = i & 1;
        const int  k0   = i * 64;
        const bool more = (i + 1) < NCHUNK;   // scalar-uniform branch

        // (1) issue next chunk's global loads into registers; their wait
        //     lands after the WMMA work below, hiding HBM latency.
        float2 nxt[TRIPS];
        if (more) {
            #pragma unroll
            for (int t = 0; t < TRIPS; ++t)
                nxt[t] = *(const float2*)(aBase + (size_t)(t * RPT) * GN + (k0 + 64));
        }

        // (2) compute current chunk: two 32-K WMMA steps, two M tiles each.
        #pragma unroll
        for (int s = 0; s < 2; ++s) {
            // A fragment, contiguous K: VGPR v<4 -> pairs {0..3}+8*half,
            //                            v>=4    -> pairs {4..7}+8*half
            union { v16bf v; uint4 q[2]; } a0, a1;
            a0.q[0] = *(const uint4*)&A_lds[cur][lm][16 * s + 8 * half];
            a0.q[1] = *(const uint4*)&A_lds[cur][lm][16 * s + 4 + 8 * half];
            a1.q[0] = *(const uint4*)&A_lds[cur][16 + lm][16 * s + 8 * half];
            a1.q[1] = *(const uint4*)&A_lds[cur][16 + lm][16 * s + 4 + 8 * half];
            // B fragment: lane holds 16 contiguous K of its column (K-major)
            const v16bf bf = *(const v16bf*)(bBase + k0 + 32 * s);
            acc0 = __builtin_amdgcn_wmma_f32_16x16x32_bf16(
                false, a0.v, false, bf, (short)0, acc0, false, false);
            acc1 = __builtin_amdgcn_wmma_f32_16x16x32_bf16(
                false, a1.v, false, bf, (short)0, acc1, false, false);
        }

        // (3) convert + store next chunk into the other buffer.
        if (more) {
            #pragma unroll
            for (int t = 0; t < TRIPS; ++t)
                A_lds[cur ^ 1][sm + t * RPT][sp] = pack2_bf16(nxt[t].x, nxt[t].y);
        }
        __syncthreads();   // one barrier per 64-K chunk
    }

    // C/D layout: VGPR r -> row r + 8*half, col = lane%16
    #pragma unroll
    for (int r = 0; r < 8; ++r) {
        out[(size_t)(m0 + r + 8 * half) * NC + col]      = acc0[r];
        out[(size_t)(m0 + 16 + r + 8 * half) * NC + col] = acc1[r];
    }
}

// ---------------------------------------------------------------------------
// Kernel 3: x2T[h][n] = bf16( relu(agg1 @ We2 + be2) ), transposed.
// block = 256 = 32 nodes x 8 h-groups (8 h each). grid = 256.
// ---------------------------------------------------------------------------
__global__ __launch_bounds__(256) void dense2_kernel(
    const float* __restrict__ agg1, const float* __restrict__ We2,
    const float* __restrict__ be2, __bf16* __restrict__ x2T)
{
    __shared__ float sW[GH1 * GH2];      // 32 KB
    __shared__ float sB[GH2];
    __shared__ float arow[32][GH1 + 1];  // 16.5 KB
    const int tid = threadIdx.x;
    for (int i = tid; i < GH1 * GH2; i += 256) sW[i] = We2[i];
    for (int i = tid; i < GH2; i += 256) sB[i] = be2[i];
    for (int i = tid; i < 32 * GH1; i += 256)
        arow[i >> 7][i & 127] = agg1[(size_t)(blockIdx.x * 32 + (i >> 7)) * GH1 + (i & 127)];
    __syncthreads();

    const int lane = tid & 31;
    const int grp  = tid >> 5;
    const int n    = blockIdx.x * 32 + lane;
    #pragma unroll 2
    for (int hh = 0; hh < 8; ++hh) {
        const int h = grp * 8 + hh;
        float s = sB[h];
        #pragma unroll 8
        for (int f = 0; f < GH1; ++f) s = fmaf(arow[lane][f], sW[f * GH2 + h], s);
        s = fmaxf(s, 0.0f);
        x2T[(size_t)h * GN + n] = (__bf16)s;
    }
}

// ---------------------------------------------------------------------------
// Kernel 5: per-node head. One wave per node:
//   emb = agg2[n]@Weo+beo ; h=[emb,act] ; h1=relu(h@Wc1+b) ; h2=relu(h1@Wc2+b)
//   q[n] = h2@Wq + bq
// Weo+Wc1 in LDS (56KB, under the safe static limit); Wc2 from L2 (hot); Wq in LDS.
// Vectors live in registers, broadcast with __shfl (wave32).
// ---------------------------------------------------------------------------
__global__ __launch_bounds__(256) void head_kernel(
    const float* __restrict__ agg2, const float* __restrict__ actions,
    const float* __restrict__ Weo, const float* __restrict__ beo,
    const float* __restrict__ Wc1, const float* __restrict__ bc1,
    const float* __restrict__ Wc2, const float* __restrict__ bc2,
    const float* __restrict__ Wq,  const float* __restrict__ bq,
    float* __restrict__ q)
{
    __shared__ float sWeo[GH2 * GH2];         // 16 KB
    __shared__ float sWc1[(GH2 + GA) * GH1];  // 40 KB
    __shared__ float sbeo[GH2], sbc1[GH1], sbc2[GH2], sWq[GH2];
    const int tid = threadIdx.x;
    for (int i = tid; i < GH2 * GH2; i += 256) sWeo[i] = Weo[i];
    for (int i = tid; i < (GH2 + GA) * GH1; i += 256) sWc1[i] = Wc1[i];
    for (int i = tid; i < GH2; i += 256) { sbeo[i] = beo[i]; sbc2[i] = bc2[i]; sWq[i] = Wq[i]; }
    for (int i = tid; i < GH1; i += 256) sbc1[i] = bc1[i];
    __syncthreads();

    const int wave = tid >> 5, lane = tid & 31;
    // 256 blocks * 8 waves = 2048 node-slots; uniform 4-trip loop (8192/2048).
    for (int n = blockIdx.x * 8 + wave; n < GN; n += 2048) {
        const float e0 = agg2[(size_t)n * GH2 + lane];
        const float e1 = agg2[(size_t)n * GH2 + 32 + lane];
        // emb = e @ Weo + beo    (j = lane, lane+32)
        float m0 = sbeo[lane], m1 = sbeo[lane + 32];
        #pragma unroll 8
        for (int i = 0; i < GH2; ++i) {
            const float vi = __shfl((i < 32) ? e0 : e1, i & 31, 32);
            m0 = fmaf(vi, sWeo[i * GH2 + lane], m0);
            m1 = fmaf(vi, sWeo[i * GH2 + lane + 32], m1);
        }
        // concat actions in lanes 0-15
        const float a0 = (lane < GA) ? actions[(size_t)n * GA + lane] : 0.0f;
        // h1 = relu([emb,act] @ Wc1 + bc1)   (j = lane + 32k, k=0..3)
        float h0 = sbc1[lane], h1 = sbc1[lane + 32], h2 = sbc1[lane + 64], h3 = sbc1[lane + 96];
        #pragma unroll 8
        for (int i = 0; i < GH2 + GA; ++i) {
            const float src = (i < 32) ? m0 : ((i < 64) ? m1 : a0);
            const float vi = __shfl(src, i & 31, 32);
            h0 = fmaf(vi, sWc1[i * GH1 + lane],      h0);
            h1 = fmaf(vi, sWc1[i * GH1 + lane + 32], h1);
            h2 = fmaf(vi, sWc1[i * GH1 + lane + 64], h2);
            h3 = fmaf(vi, sWc1[i * GH1 + lane + 96], h3);
        }
        h0 = fmaxf(h0, 0.0f); h1 = fmaxf(h1, 0.0f);
        h2 = fmaxf(h2, 0.0f); h3 = fmaxf(h3, 0.0f);
        // h2l = relu(h1 @ Wc2 + bc2)
        float g0 = sbc2[lane], g1 = sbc2[lane + 32];
        #pragma unroll 8
        for (int i = 0; i < GH1; ++i) {
            const float src = (i < 32) ? h0 : ((i < 64) ? h1 : ((i < 96) ? h2 : h3));
            const float vi = __shfl(src, i & 31, 32);
            g0 = fmaf(vi, Wc2[i * GH2 + lane],      g0);
            g1 = fmaf(vi, Wc2[i * GH2 + lane + 32], g1);
        }
        g0 = fmaxf(g0, 0.0f); g1 = fmaxf(g1, 0.0f);
        // q = g @ Wq + bq, wave reduction
        float p = fmaf(g0, sWq[lane], g1 * sWq[lane + 32]);
        #pragma unroll
        for (int off = 16; off > 0; off >>= 1) p += __shfl_down(p, off, 32);
        if (lane == 0) q[n] = p + bq[0];
    }
}

// ---------------------------------------------------------------------------
// Launch. Workspace layout (9 MB):
//   [0,2MB)   x1T  bf16 [H1][N]
//   [2,6MB)   agg1 f32  [N][H1]
//   [6,7MB)   x2T  bf16 [H2][N]
//   [7,9MB)   agg2 f32  [N][H2]
// Input order: nodes, edges(unused, as in reference), adjacency, actions,
//              We1,be1, We2,be2, Weo,beo, Wc1,bc1, Wc2,bc2, Wq,bq
// ---------------------------------------------------------------------------
extern "C" void kernel_launch(void* const* d_in, const int* in_sizes, int n_in,
                              void* d_out, int out_size, void* d_ws, size_t ws_size,
                              hipStream_t stream) {
    const float* nodes   = (const float*)d_in[0];
    const float* adj     = (const float*)d_in[2];
    const float* actions = (const float*)d_in[3];
    const float* We1 = (const float*)d_in[4];  const float* be1 = (const float*)d_in[5];
    const float* We2 = (const float*)d_in[6];  const float* be2 = (const float*)d_in[7];
    const float* Weo = (const float*)d_in[8];  const float* beo = (const float*)d_in[9];
    const float* Wc1 = (const float*)d_in[10]; const float* bc1 = (const float*)d_in[11];
    const float* Wc2 = (const float*)d_in[12]; const float* bc2 = (const float*)d_in[13];
    const float* Wq  = (const float*)d_in[14]; const float* bq  = (const float*)d_in[15];
    float* q = (float*)d_out;

    char* ws = (char*)d_ws;
    __bf16* x1T  = (__bf16*)(ws);
    float*  agg1 = (float*)(ws + (size_t)2 * 1024 * 1024);
    __bf16* x2T  = (__bf16*)(ws + (size_t)6 * 1024 * 1024);
    float*  agg2 = (float*)(ws + (size_t)7 * 1024 * 1024);

    dense1_kernel<<<GN / 32, 256, 0, stream>>>(nodes, We1, be1, x1T);
    agg_kernel<GH1><<<GN / 32, 256, 0, stream>>>(adj, x1T, agg1);
    dense2_kernel<<<GN / 32, 256, 0, stream>>>(agg1, We2, be2, x2T);
    agg_kernel<GH2><<<GN / 32, 128, 0, stream>>>(adj, x2T, agg2);
    head_kernel<<<256, 256, 0, stream>>>(agg2, actions, Weo, beo, Wc1, bc1,
                                         Wc2, bc2, Wq, bq, q);
}
